// NormalEstimation_7421703488127
// MI455X (gfx1250) — matrix operand order repaired and checked by reading, not compiled, not run on hardware
//
#include <hip/hip_runtime.h>
#include <math.h>

typedef __attribute__((ext_vector_type(16))) _Float16 v16h;
typedef __attribute__((ext_vector_type(8)))  _Float16 v8h;
typedef __attribute__((ext_vector_type(2)))  _Float16 v2h;
typedef __attribute__((ext_vector_type(8)))  float    v8f;
typedef __attribute__((ext_vector_type(4)))  float    v4f;

#define NPTS 50000
#define KNBR 32
#define HDIM 64
#define ROWP 72          // padded abuf row stride in halfs (144B = 36 banks; gcd(36,64)=4)
#define RSP  20          // padded rs row stride in floats (80B: 16B-aligned, halves on disjoint banks)
#define EPSV 1e-8f
#define WAVES_PER_BLOCK 4

__global__ __launch_bounds__(128)
void gnn_normal_kernel(const float* __restrict__ old_weights, // N*K
                       const float* __restrict__ pos,         // N*3
                       const float* __restrict__ normals,     // N*3
                       const int*   __restrict__ dense_l,     // N*K
                       const float* __restrict__ stddev,      // 1
                       const float* __restrict__ W1,          // 6*64
                       const float* __restrict__ b1,          // 64
                       const float* __restrict__ W2,          // 128*64
                       const float* __restrict__ b2,          // 64
                       const float* __restrict__ W3,          // 64
                       const float* __restrict__ b3,          // 1
                       float* __restrict__ out_normals,       // N*3
                       float* __restrict__ out_weights)       // N*K
{
    // W2 transposed to [n][k] so each B-fragment is one contiguous 32B per lane.
    __shared__ alignas(64) _Float16 s_w2t[HDIM][2 * HDIM];                 // 16 KB
    __shared__ alignas(64) _Float16 s_abuf[WAVES_PER_BLOCK][KNBR][ROWP];   // 18 KB (padded rows)
    __shared__ alignas(64) _Float16 s_agg[WAVES_PER_BLOCK][2 * HDIM];      // 1 KB (agg stored once)
    __shared__ alignas(64) float s_rs[WAVES_PER_BLOCK][KNBR][RSP];         // 10 KB (rs transpose)
    __shared__ alignas(16) float s_w1[6 * HDIM];
    __shared__ alignas(16) float s_b1[HDIM];
    __shared__ alignas(16) float s_b2[HDIM];
    __shared__ alignas(16) float s_w3[HDIM];

    const int tid = threadIdx.x;

    // ---- cooperative LDS init (weights shared by every point) ----
    for (int idx = tid; idx < 2 * HDIM * HDIM; idx += blockDim.x) {
        const int k = idx >> 6;       // 0..127
        const int c = idx & 63;       // 0..63
        s_w2t[c][k] = (_Float16)W2[idx];
    }
    for (int idx = tid; idx < 6 * HDIM; idx += blockDim.x) s_w1[idx] = W1[idx];
    for (int idx = tid; idx < HDIM; idx += blockDim.x) {
        s_b1[idx] = b1[idx];
        s_b2[idx] = b2[idx];
        s_w3[idx] = W3[idx];
    }
    __syncthreads();

    const int wave  = tid >> 5;
    const int lane  = tid & 31;
    const int half  = lane >> 4;   // 0: lanes 0-15, 1: lanes 16-31
    const int l16   = lane & 15;
    const int gwave = blockIdx.x * WAVES_PER_BLOCK + wave;
    const int nwav  = gridDim.x * WAVES_PER_BLOCK;

    const float inv_std = 1.0f / stddev[0];
    const float bias3   = b3[0];

    _Float16 (* __restrict__ abuf)[ROWP] = s_abuf[wave];
    _Float16* __restrict__ aggw = s_agg[wave];
    float (* __restrict__ rsb)[RSP] = s_rs[wave];

    // ---- hoist loop-invariant B-fragments (W2) + per-lane b2/W3 into registers ----
    // B layout used here (same k order as A): lane column n = lane%16; lanes 0-15 hold K 0-15,
    // lanes 16-31 hold K 16-31, K contiguous -> one 32B LDS read per fragment.
    v16h  bfrag[16];
    float binit[4], w3v[4];
    #pragma unroll
    for (int nt = 0; nt < 4; ++nt) {
        binit[nt] = s_b2[nt * 16 + l16];
        w3v[nt]   = s_w3[nt * 16 + l16];
        #pragma unroll
        for (int kc = 0; kc < 4; ++kc)
            bfrag[nt * 4 + kc] = *(const v16h*)&s_w2t[nt * 16 + l16][kc * 32 + half * 16];
    }

    for (int n = gwave; n < NPTS; n += nwav) {
        // ---------- edge features: lane = neighbor k ----------
        const int   j  = dense_l[n * KNBR + lane];
        const float px = pos[3 * n + 0], py = pos[3 * n + 1], pz = pos[3 * n + 2];
        const float rx = pos[3 * j + 0] - px;
        const float ry = pos[3 * j + 1] - py;
        const float rz = pos[3 * j + 2] - pz;
        const float dist = sqrtf(rx * rx + ry * ry + rz * rz);
        const float nx = normals[3 * n + 0], ny = normals[3 * n + 1], nz = normals[3 * n + 2];
        const float ndot = fabsf(nx * rx + ny * ry + nz * rz) / (dist + EPSV);
        const float f0 = rx * inv_std, f1 = ry * inv_std, f2 = rz * inv_std;
        const float f3 = dist * inv_std, f4 = ndot;
        const float f5 = old_weights[n * KNBR + lane];

        // ---------- layer 1: h = relu(feat@W1+b1), store row (=neighbor) to LDS ----------
        #pragma unroll
        for (int gg = 0; gg < 8; ++gg) {           // 8 channels per 16B store
            v8h pk;
            #pragma unroll
            for (int s = 0; s < 2; ++s) {          // 4 channels per float4 group
                const int cb = gg * 8 + s * 4;
                const v4f bb = *(const v4f*)&s_b1[cb];
                const v4f wa = *(const v4f*)&s_w1[0 * HDIM + cb];
                const v4f wb = *(const v4f*)&s_w1[1 * HDIM + cb];
                const v4f wc = *(const v4f*)&s_w1[2 * HDIM + cb];
                const v4f wd = *(const v4f*)&s_w1[3 * HDIM + cb];
                const v4f we = *(const v4f*)&s_w1[4 * HDIM + cb];
                const v4f wf = *(const v4f*)&s_w1[5 * HDIM + cb];
                #pragma unroll
                for (int e = 0; e < 4; ++e) {
                    float a = bb[e];
                    a = fmaf(f0, wa[e], a);
                    a = fmaf(f1, wb[e], a);
                    a = fmaf(f2, wc[e], a);
                    a = fmaf(f3, wd[e], a);
                    a = fmaf(f4, we[e], a);
                    a = fmaf(f5, wf[e], a);
                    pk[s * 4 + e] = (_Float16)fmaxf(a, 0.0f);
                }
            }
            *(v8h*)&abuf[lane][gg * 8] = pk;
        }
        __builtin_amdgcn_wave_barrier(); // same-wave LDS is in-order

        // ---------- agg = column max via LDS transpose-gather (lane owns 2 columns) ----------
        // Padded row stride (36 banks/row) makes the 32-row walk conflict-free.
        {
            v2h cm = *(const v2h*)&abuf[0][2 * lane];
            #pragma unroll
            for (int r2 = 1; r2 < KNBR; ++r2) {
                const v2h x = *(const v2h*)&abuf[r2][2 * lane];
                cm = __builtin_elementwise_max(cm, x);   // v_pk_max_num_f16
            }
            // store once; A columns 64..127 are identical for every row
            *(v2h*)&aggw[2 * lane] = cm;
        }
        __builtin_amdgcn_wave_barrier();

        // ---------- preload the 6 unique A-fragments ----------
        // lanes 0-15: row=l16, K{0..15}; lanes 16-31: row=l16, K{16..31}
        v16h afr[4];   // [mt*2+kc], kc<2 (h part)
        #pragma unroll
        for (int mt = 0; mt < 2; ++mt) {
            #pragma unroll
            for (int kc = 0; kc < 2; ++kc) {
                const _Float16* rp = &abuf[mt * 16 + l16][kc * 32 + half * 16];
                const v8h lo = *(const v8h*)(rp);
                const v8h hi = *(const v8h*)(rp + 8);
                #pragma unroll
                for (int e = 0; e < 8; ++e) { afr[mt * 2 + kc][e] = lo[e]; afr[mt * 2 + kc][e + 8] = hi[e]; }
            }
        }
        v16h gfr[2];   // agg part, shared by both M-tiles (broadcast LDS reads)
        #pragma unroll
        for (int kc = 0; kc < 2; ++kc) {
            const _Float16* rp = &aggw[kc * 32 + half * 16];
            const v8h lo = *(const v8h*)(rp);
            const v8h hi = *(const v8h*)(rp + 8);
            #pragma unroll
            for (int e = 0; e < 8; ++e) { gfr[kc][e] = lo[e]; gfr[kc][e + 8] = hi[e]; }
        }

        // ---------- layer 2: (32x128)@(128x64) via v_wmma_f32_16x16x32_f16 ----------
        #pragma unroll
        for (int mt = 0; mt < 2; ++mt) {
            float rs[8];
            #pragma unroll
            for (int r = 0; r < 8; ++r) rs[r] = 0.0f;

            #pragma unroll
            for (int nt = 0; nt < 4; ++nt) {
                v8f acc;
                #pragma unroll
                for (int r = 0; r < 8; ++r) acc[r] = binit[nt]; // C layout: N = lane%16
                acc = __builtin_amdgcn_wmma_f32_16x16x32_f16(
                    false, afr[mt * 2 + 0], false, bfrag[nt * 4 + 0], (short)0, acc, false, false);
                acc = __builtin_amdgcn_wmma_f32_16x16x32_f16(
                    false, afr[mt * 2 + 1], false, bfrag[nt * 4 + 1], (short)0, acc, false, false);
                acc = __builtin_amdgcn_wmma_f32_16x16x32_f16(
                    false, gfr[0],          false, bfrag[nt * 4 + 2], (short)0, acc, false, false);
                acc = __builtin_amdgcn_wmma_f32_16x16x32_f16(
                    false, gfr[1],          false, bfrag[nt * 4 + 3], (short)0, acc, false, false);
                #pragma unroll
                for (int r = 0; r < 8; ++r)
                    rs[r] = fmaf(fmaxf(acc[r], 0.0f), w3v[nt], rs[r]); // relu + @W3 fold
            }
            // scatter partial row-sums: row m = mt*16 + half*8 + r, column = l16
            #pragma unroll
            for (int r = 0; r < 8; ++r)
                rsb[mt * 16 + half * 8 + r][l16] = rs[r];
        }
        __builtin_amdgcn_wave_barrier();

        // ---------- per-lane row-sum gather: lane owns its own edge's sigmoid ----------
        const v4f q0 = *(const v4f*)&rsb[lane][0];
        const v4f q1 = *(const v4f*)&rsb[lane][4];
        const v4f q2 = *(const v4f*)&rsb[lane][8];
        const v4f q3 = *(const v4f*)&rsb[lane][12];
        const v4f qs = (q0 + q1) + (q2 + q3);
        const float tot = (qs[0] + qs[1]) + (qs[2] + qs[3]);
        const float wk  = 1.0f / (1.0f + __expf(-(tot + bias3)));
        out_weights[n * KNBR + lane] = wk;

        // ---------- weighted covariance (wave reduction) ----------
        float cxx = wk * rx * rx, cxy = wk * rx * ry, cxz = wk * rx * rz;
        float cyy = wk * ry * ry, cyz = wk * ry * rz, czz = wk * rz * rz;
        #pragma unroll
        for (int off = 16; off >= 1; off >>= 1) {
            cxx += __shfl_xor(cxx, off, 32);
            cxy += __shfl_xor(cxy, off, 32);
            cxz += __shfl_xor(cxz, off, 32);
            cyy += __shfl_xor(cyy, off, 32);
            cyz += __shfl_xor(cyz, off, 32);
            czz += __shfl_xor(czz, off, 32);
        }

        // ---------- lane 0: closed-form 3x3 symmetric eigensolve ----------
        if (lane == 0) {
            const float a00 = cxx, a01 = cxy, a02 = cxz;
            const float a11 = cyy, a12 = cyz, a22 = czz;
            const float q  = (a00 + a11 + a22) * (1.0f / 3.0f);
            const float p1 = a01 * a01 + a02 * a02 + a12 * a12;
            const float d0 = a00 - q, d1 = a11 - q, d2 = a22 - q;
            const float p2 = d0 * d0 + d1 * d1 + d2 * d2 + 2.0f * p1;
            const float p  = sqrtf(p2 * (1.0f / 6.0f));
            float e0, e1, e2;
            if (p < 1e-20f) {
                e0 = e1 = e2 = q;
            } else {
                const float ip  = 1.0f / p;
                const float b00 = d0 * ip, b01 = a01 * ip, b02 = a02 * ip;
                const float b11 = d1 * ip, b12 = a12 * ip, b22 = d2 * ip;
                float rdet = 0.5f * (b00 * (b11 * b22 - b12 * b12)
                                   - b01 * (b01 * b22 - b12 * b02)
                                   + b02 * (b01 * b12 - b11 * b02));
                rdet = fminf(1.0f, fmaxf(-1.0f, rdet));
                const float phi = acosf(rdet) * (1.0f / 3.0f);
                e0 = q + 2.0f * p * cosf(phi);
                e2 = q + 2.0f * p * cosf(phi + 2.09439510239f);
                e1 = 3.0f * q - e0 - e2;
            }
            float lam = e0;
            if (fabsf(e1) < fabsf(lam)) lam = e1;
            if (fabsf(e2) < fabsf(lam)) lam = e2;

            // eigenvector of min-|lambda|: cross products of rows of (A - lam I)
            const float r0x = a00 - lam, r0y = a01,       r0z = a02;
            const float r1x = a01,       r1y = a11 - lam, r1z = a12;
            const float r2x = a02,       r2y = a12,       r2z = a22 - lam;
            float v0x = r0y * r1z - r0z * r1y, v0y = r0z * r1x - r0x * r1z, v0z = r0x * r1y - r0y * r1x;
            float v1x = r0y * r2z - r0z * r2y, v1y = r0z * r2x - r0x * r2z, v1z = r0x * r2y - r0y * r2x;
            float v2x = r1y * r2z - r1z * r2y, v2y = r1z * r2x - r1x * r2z, v2z = r1x * r2y - r1y * r2x;
            const float n0 = v0x * v0x + v0y * v0y + v0z * v0z;
            const float n1 = v1x * v1x + v1y * v1y + v1z * v1z;
            const float n2 = v2x * v2x + v2y * v2y + v2z * v2z;
            float vx = v0x, vy = v0y, vz = v0z, nn = n0;
            if (n1 > nn) { vx = v1x; vy = v1y; vz = v1z; nn = n1; }
            if (n2 > nn) { vx = v2x; vy = v2y; vz = v2z; nn = n2; }
            if (nn < 1e-30f) { vx = 0.0f; vy = 0.0f; vz = 1.0f; nn = 1.0f; }
            const float inv = 1.0f / sqrtf(nn);
            out_normals[3 * n + 0] = vx * inv;
            out_normals[3 * n + 1] = vy * inv;
            out_normals[3 * n + 2] = vz * inv;
        }
    }
}

extern "C" void kernel_launch(void* const* d_in, const int* in_sizes, int n_in,
                              void* d_out, int out_size, void* d_ws, size_t ws_size,
                              hipStream_t stream) {
    (void)in_sizes; (void)n_in; (void)out_size; (void)d_ws; (void)ws_size;
    const float* old_weights = (const float*)d_in[0];
    const float* pos         = (const float*)d_in[1];
    const float* normals     = (const float*)d_in[2];
    /* d_in[3] = edge_idx_l (unused; dense_l carries the same info) */
    const int*   dense_l     = (const int*)  d_in[4];
    const float* stddev      = (const float*)d_in[5];
    const float* W1          = (const float*)d_in[6];
    const float* b1          = (const float*)d_in[7];
    const float* W2          = (const float*)d_in[8];
    const float* b2          = (const float*)d_in[9];
    const float* W3          = (const float*)d_in[10];
    const float* b3          = (const float*)d_in[11];

    float* out_normals = (float*)d_out;            // N*3 first
    float* out_weights = out_normals + NPTS * 3;   // then N*K

    const int blocks = (NPTS + WAVES_PER_BLOCK - 1) / WAVES_PER_BLOCK; // 1 point per wave
    gnn_normal_kernel<<<blocks, WAVES_PER_BLOCK * 32, 0, stream>>>(
        old_weights, pos, normals, dense_l, stddev,
        W1, b1, W2, b2, W3, b3, out_normals, out_weights);
}
